// MutualInformation_21844203667943
// MI455X (gfx1250) — matrix operand order repaired and checked by reading, not compiled, not run on hardware
//
#include <hip/hip_runtime.h>

// MI(x,y): J = (1/B) x^T y ; mi = sum J * log((J+eps)/(px py + eps))
// x,y: [B=8192, D=2048] f32.  GEMM M=N=2048, K=8192 via bf16x3 WMMA split.

#define MI_EPS 1e-9f
#define BSZ 8192
#define DIM 2048
#define BM  128
#define BN  128
#define BK  32
#define LDA 40          // padded K-stride (bf16 elems) -> 80B rows, 16B aligned

typedef __attribute__((ext_vector_type(16))) __bf16 v16bf;
typedef __attribute__((ext_vector_type(8)))  __bf16 v8bf;
typedef __attribute__((ext_vector_type(2)))  __bf16 v2bf;
typedef __attribute__((ext_vector_type(8)))  float  v8f;
typedef __attribute__((ext_vector_type(4)))  float  v4f;

// ---------------- marginals: px = mean(x,0), py = mean(y,0); zero d_out ----
__global__ void mi_means_kernel(const float* __restrict__ x,
                                const float* __restrict__ y,
                                float* __restrict__ marg,
                                float* __restrict__ out) {
  const int bid = blockIdx.x;                 // 0..7 -> x cols, 8..15 -> y cols
  const float* __restrict__ src = (bid < 8) ? x : y;
  const int col = (bid & 7) * 256 + threadIdx.x;
  const float* p = src + col;
  float s0 = 0.f, s1 = 0.f, s2 = 0.f, s3 = 0.f;
  for (int r = 0; r < BSZ; r += 4) {
    s0 += p[(size_t)(r + 0) * DIM];
    s1 += p[(size_t)(r + 1) * DIM];
    s2 += p[(size_t)(r + 2) * DIM];
    s3 += p[(size_t)(r + 3) * DIM];
  }
  marg[((bid < 8) ? 0 : DIM) + col] = ((s0 + s1) + (s2 + s3)) * (1.0f / BSZ);
  if (bid == 0 && threadIdx.x == 0) out[0] = 0.0f;
}

// Build a 16-elem bf16 fragment (8 VGPRs) from two 16B LDS reads.
__device__ __forceinline__ v16bf ldsFrag(const __bf16* row, int o0, int o1) {
  v8bf a = *(const v8bf*)(row + o0);
  v8bf b = *(const v8bf*)(row + o1);
  return __builtin_shufflevector(a, b, 0, 1, 2, 3, 4, 5, 6, 7,
                                       8, 9, 10, 11, 12, 13, 14, 15);
}

// Split two f32 (adjacent K rows, same column) into bf16 hi/lo and store as
// packed 32-bit DS writes: [col][r0],[col][r0+1] are contiguous in LDS.
__device__ __forceinline__ void stagePair(__bf16* __restrict__ hiP,
                                          __bf16* __restrict__ loP,
                                          v4f a, v4f b) {
#pragma unroll
  for (int e = 0; e < 4; ++e) {
    const float fa = a[e], fb = b[e];
    const __bf16 h0 = (__bf16)fa;              // v_cvt_pk_bf16_f32 (both slots)
    const __bf16 h1 = (__bf16)fb;
    const __bf16 l0 = (__bf16)(fa - (float)h0);
    const __bf16 l1 = (__bf16)(fb - (float)h1);
    *(v2bf*)(hiP + e * LDA) = (v2bf){h0, h1};  // ds_store_b32
    *(v2bf*)(loP + e * LDA) = (v2bf){l0, l1};  // ds_store_b32
  }
}

// ---------------- fused GEMM (bf16x3) + MI reduction -----------------------
__global__ __launch_bounds__(256)
void mi_gemm_kernel(const float* __restrict__ x, const float* __restrict__ y,
                    const float* __restrict__ marg, float* __restrict__ out) {
  __shared__ __align__(16) __bf16 Ahi[BM][LDA];
  __shared__ __align__(16) __bf16 Alo[BM][LDA];
  __shared__ __align__(16) __bf16 Bhi[BN][LDA];
  __shared__ __align__(16) __bf16 Blo[BN][LDA];
  __shared__ float red[8];

  const int tid  = threadIdx.x;
  const int lane = tid & 31;
  const int w    = tid >> 5;        // 8 waves
  const int wm   = w & 1;           // 2 waves along M (64 rows each)
  const int wn   = w >> 1;          // 4 waves along N (32 cols each)
  const int ml   = lane & 15;       // row-in-tile (A: M, B: N, C: N)
  const int h    = lane >> 4;       // lane half -> K half / M+8
  const int m0   = blockIdx.x * BM;
  const int n0   = blockIdx.y * BN;

  // loader mapping: 2 pair-units/thread/tile; each unit = 2 rows x 4 cols
  int pr0[2], pc[2];
#pragma unroll
  for (int q = 0; q < 2; ++q) {
    const int p = tid + q * 256;               // 0..511
    pr0[q] = (p >> 5) * 2;                     // even K row 0..30
    pc[q]  = (p & 31) * 4;                     // col base 0..124
  }

  v8f acc[4][2];
#pragma unroll
  for (int i = 0; i < 4; ++i)
#pragma unroll
    for (int j = 0; j < 2; ++j)
      acc[i][j] = (v8f){0.f, 0.f, 0.f, 0.f, 0.f, 0.f, 0.f, 0.f};

  // register prefetch buffers for the current K tile
  v4f xr[2][2], yr[2][2];
#pragma unroll
  for (int q = 0; q < 2; ++q) {
    xr[q][0] = *(const v4f*)(x + (size_t)(pr0[q] + 0) * DIM + m0 + pc[q]);
    xr[q][1] = *(const v4f*)(x + (size_t)(pr0[q] + 1) * DIM + m0 + pc[q]);
    yr[q][0] = *(const v4f*)(y + (size_t)(pr0[q] + 0) * DIM + n0 + pc[q]);
    yr[q][1] = *(const v4f*)(y + (size_t)(pr0[q] + 1) * DIM + n0 + pc[q]);
  }

#pragma unroll 1
  for (int k0 = 0; k0 < BSZ; k0 += BK) {
    // ---- convert + store the prefetched tile into LDS ---------------------
#pragma unroll
    for (int q = 0; q < 2; ++q) {
      stagePair(&Ahi[pc[q]][pr0[q]], &Alo[pc[q]][pr0[q]], xr[q][0], xr[q][1]);
      stagePair(&Bhi[pc[q]][pr0[q]], &Blo[pc[q]][pr0[q]], yr[q][0], yr[q][1]);
    }
    __syncthreads();

    // ---- issue global loads for the NEXT K tile (hidden under WMMAs) ------
    if (k0 + BK < BSZ) {
      const int kn = k0 + BK;
#pragma unroll
      for (int q = 0; q < 2; ++q) {
        xr[q][0] = *(const v4f*)(x + (size_t)(kn + pr0[q] + 0) * DIM + m0 + pc[q]);
        xr[q][1] = *(const v4f*)(x + (size_t)(kn + pr0[q] + 1) * DIM + m0 + pc[q]);
        yr[q][0] = *(const v4f*)(y + (size_t)(kn + pr0[q] + 0) * DIM + n0 + pc[q]);
        yr[q][1] = *(const v4f*)(y + (size_t)(kn + pr0[q] + 1) * DIM + n0 + pc[q]);
      }
    }

    // ---- fragments per CDNA5 16-bit WMMA layouts --------------------------
    // A 16x32: V0-3 = K[h*8 .. +7], V4-7 = K[16+h*8 ..]
    // B 32x16: V0-3 = K[h*8 .. +7], V4-7 = K[16+h*8 ..]
    v16bf ah[4], al[4], bh[2], bl[2];
#pragma unroll
    for (int t = 0; t < 4; ++t) {
      const int ar = wm * 64 + t * 16 + ml;
      ah[t] = ldsFrag(&Ahi[ar][0], h * 8, 16 + h * 8);
      al[t] = ldsFrag(&Alo[ar][0], h * 8, 16 + h * 8);
    }
#pragma unroll
    for (int t = 0; t < 2; ++t) {
      const int br = wn * 32 + t * 16 + ml;
      bh[t] = ldsFrag(&Bhi[br][0], h * 8, 16 + h * 8);
      bl[t] = ldsFrag(&Blo[br][0], h * 8, 16 + h * 8);
    }

#pragma unroll
    for (int tm = 0; tm < 4; ++tm)
#pragma unroll
      for (int tn = 0; tn < 2; ++tn) {
        acc[tm][tn] = __builtin_amdgcn_wmma_f32_16x16x32_bf16(
            false, ah[tm], false, bh[tn], (short)0, acc[tm][tn], false, false);
        acc[tm][tn] = __builtin_amdgcn_wmma_f32_16x16x32_bf16(
            false, ah[tm], false, bl[tn], (short)0, acc[tm][tn], false, false);
        acc[tm][tn] = __builtin_amdgcn_wmma_f32_16x16x32_bf16(
            false, al[tm], false, bh[tn], (short)0, acc[tm][tn], false, false);
      }
    __syncthreads();   // protect LDS before next iteration's staging
  }

  // ---- fused MI epilogue: C layout lane->N, VGPR r -> M=r (+8 high half) --
  const float* px = marg;
  const float* py = marg + DIM;
  float msum = 0.f;
#pragma unroll
  for (int tm = 0; tm < 4; ++tm)
#pragma unroll
    for (int tn = 0; tn < 2; ++tn) {
      const int gj = n0 + wn * 32 + tn * 16 + ml;
      const float pj = py[gj];
#pragma unroll
      for (int r = 0; r < 8; ++r) {
        const int gi = m0 + wm * 64 + tm * 16 + r + h * 8;
        const float pi = px[gi];
        const float jv = acc[tm][tn][r] * (1.0f / BSZ);
        msum += jv * __logf((jv + MI_EPS) / (pi * pj + MI_EPS));
      }
    }
  // wave32 reduce
#pragma unroll
  for (int off = 16; off > 0; off >>= 1)
    msum += __shfl_down(msum, off, 32);
  if (lane == 0) red[w] = msum;
  __syncthreads();
  if (tid == 0) {
    float t = 0.f;
#pragma unroll
    for (int i = 0; i < 8; ++i) t += red[i];
    atomicAdd(out, t);
  }
}

extern "C" void kernel_launch(void* const* d_in, const int* in_sizes, int n_in,
                              void* d_out, int out_size, void* d_ws, size_t ws_size,
                              hipStream_t stream) {
  const float* x = (const float*)d_in[0];
  const float* y = (const float*)d_in[1];
  float* out  = (float*)d_out;
  float* marg = (float*)d_ws;                 // px[2048] then py[2048]

  mi_means_kernel<<<16, 256, 0, stream>>>(x, y, marg, out);

  dim3 grid(DIM / BM, DIM / BN);              // 16 x 16 = 256 workgroups
  mi_gemm_kernel<<<grid, 256, 0, stream>>>(x, y, marg, out);
}
